// VN_DGCNN_partseg_class_48704929137387
// MI455X (gfx1250) — matrix operand order repaired and checked by reading, not compiled, not run on hardware
//
#include <hip/hip_runtime.h>
#include <hip/hip_bf16.h>

// ---------------------------------------------------------------------------
// VN-DGCNN partseg forward for gfx1250 (MI455X). All channel contractions run
// through v_wmma_f32_16x16x32_f16 (f16 in, f32 accumulate); elementwise VN
// math stays in VALU kernels.
//
// GEMM hot loop: unconditional clamped loads, no data masking at all
// (clamped rows/cols only affect C entries that are never stored); K-tail
// zeroes only the B fragment (zero B element kills the product for k>=Kd).
// Steady state = 2 pointer bumps + 32 batched global_load_b32 + cvt + WMMA.
// ---------------------------------------------------------------------------

typedef __attribute__((ext_vector_type(16))) _Float16 v16h;
typedef __attribute__((ext_vector_type(8)))  float    v8f;

#define EPS_   1e-6f
#define BNEPS_ 1e-5f
#define NEG_   0.2f

static constexpr int BB   = 4;
static constexpr int NN   = 1024;
static constexpr int KNNB = 20;
static constexpr int CC   = 42;     // 128 // 3
static constexpr int C9   = 341;    // 1024 // 3
static constexpr int D2   = 682;    // 2 * C9
static constexpr int OUTC = 50;
static constexpr long long NKLL = (long long)NN * KNNB;  // 20480

// ---------------------------------------------------------------------------
// Generic strided batched GEMM: C[b,m,s] = sum_k W[b,m,k] * A[b,k,s]
// f32 memory, f16 WMMA compute, f32 accumulate.
// Per-wave 16x16 C tile; 4 waves/block cover 4 consecutive S tiles.
//
// Fragment K-index maps (contiguous K order chosen by this code, same for A and B):
//   A frag (MxK): lane l -> M = m0+(l&15); base K = k0 + (l>>4)*16;
//                 element h -> K = base + h
//   B frag (KxN): lane l -> N = s0+(l&15); base K = k0 + (l>>4)*16;
//                 element h -> K = base + h
//   C/D (16x16 f32): VGPR r -> M = m0 + r + (l>>4)*8 ; N = s0 + (l&15)
// ---------------------------------------------------------------------------
__global__ void __launch_bounds__(128)
gemm_f16_wmma(const float* __restrict__ W, long long wB, long long wM, long long wK,
              const float* __restrict__ A, long long aB, long long aK, long long aS,
              float* __restrict__ Cp, long long cB, long long cM, long long cS,
              int M, int Kd, int S)
{
    const int lane = threadIdx.x & 31;
    const int wave = threadIdx.x >> 5;
    const int s0 = (blockIdx.x * 4 + wave) * 16;
    const int m0 = blockIdx.y * 16;
    const int b  = blockIdx.z;
    if (s0 >= S) return;  // wave-uniform: EXEC stays all-ones inside WMMA region

    const int l15 = lane & 15;
    const int kh  = lane >> 4;
    const int mA  = m0 + l15;
    const int sB  = s0 + l15;
    // Clamp row/col: garbage from duplicated edge rows/cols lands only in
    // C entries whose store is masked off below, so no data masking needed.
    const int mAc = (mA < M) ? mA : (M - 1);
    const int sBc = (sB < S) ? sB : (S - 1);

    const float* Wrow = W + (long long)b * wB + (long long)mAc * wM;
    const float* Acol = A + (long long)b * aB + (long long)sBc * aS;

    const float* Wbase = Wrow + (long long)(kh * 16) * wK;
    const float* Abase = Acol + (long long)(kh * 16) * aK;

    v8f acc = {};
    int k0 = 0;

    // ---- full 32-wide K tiles: unconditional batched loads, no masking ----
    for (; k0 + 32 <= Kd; k0 += 32) {
        float wv[16], av[16];
#pragma unroll
        for (int h = 0; h < 16; ++h) {
            wv[h] = Wbase[(long long)h * wK];
            av[h] = Abase[(long long)h * aK];
        }
        v16h af, bf;
#pragma unroll
        for (int h = 0; h < 16; ++h) {
            af[h] = (_Float16)wv[h];
            bf[h] = (_Float16)av[h];
        }
        acc = __builtin_amdgcn_wmma_f32_16x16x32_f16(
            false, af, false, bf, (short)0, acc, false, false);
        Wbase += 32LL * wK;
        Abase += 32LL * aK;
    }

    // ---- tail tile: clamp K addresses; zero ONLY the B fragment for k>=Kd
    //      (zero B element nullifies the product, regardless of A garbage) ----
    if (k0 < Kd) {
        float wv[16], av[16];
#pragma unroll
        for (int h = 0; h < 16; ++h) {
            const int ka = k0 + kh * 16 + h;
            const int kb = k0 + kh * 16 + h;
            const int kac = (ka < Kd) ? ka : (Kd - 1);
            const int kbc = (kb < Kd) ? kb : (Kd - 1);
            wv[h] = Wrow[(long long)kac * wK];
            av[h] = Acol[(long long)kbc * aK];
        }
        v16h af, bf;
#pragma unroll
        for (int h = 0; h < 16; ++h) {
            const int kb = k0 + kh * 16 + h;
            af[h] = (_Float16)wv[h];
            bf[h] = (kb < Kd) ? (_Float16)av[h] : (_Float16)0.0f;
        }
        acc = __builtin_amdgcn_wmma_f32_16x16x32_f16(
            false, af, false, bf, (short)0, acc, false, false);
    }

    float* Cb = Cp + (long long)b * cB;
#pragma unroll
    for (int r = 0; r < 8; ++r) {
        const int mC = m0 + r + (kh << 3);
        const int sC = s0 + l15;
        if (mC < M && sC < S)
            Cb[(long long)mC * cM + (long long)sC * cS] = acc[r];
    }
}

// ---------------------------------------------------------------------------
// Top-20 of neg_dist row, from Gram matrix G (B,N,N).
// neg_dist[i,j] = 2 G[i,j] - G[i,i] - G[j,j]; ties -> lowest index (jax top_k).
// ---------------------------------------------------------------------------
__global__ void topk_kernel(const float* __restrict__ G, int* __restrict__ idx)
{
    const int t = blockIdx.x * blockDim.x + threadIdx.x;
    if (t >= BB * NN) return;
    const int b = t / NN, i = t - b * NN;
    const float* Gb = G + (long long)b * NN * NN;
    const float sqi = Gb[(long long)i * NN + i];

    float vals[KNNB];
    int   ids[KNNB];
#pragma unroll
    for (int k = 0; k < KNNB; ++k) { vals[k] = -3.4e38f; ids[k] = 0; }

    for (int j = 0; j < NN; ++j) {
        const float v = 2.0f * Gb[(long long)i * NN + j] - sqi - Gb[(long long)j * NN + j];
        if (v > vals[KNNB - 1]) {
            int p = KNNB - 1;
            while (p > 0 && vals[p - 1] < v) {
                vals[p] = vals[p - 1]; ids[p] = ids[p - 1]; --p;
            }
            vals[p] = v; ids[p] = j;
        }
    }
#pragma unroll
    for (int k = 0; k < KNNB; ++k)
        idx[((long long)b * NN + i) * KNNB + k] = ids[k];
}

// ---------------------------------------------------------------------------
// Graph feature: out (B, 2*Cf, 3, N, K); xf (B, 3*Cf, N).
// c<Cf: xf[b,c*3+v,idx]-xf[b,c*3+v,n] ; c>=Cf: xf[b,(c-Cf)*3+v,n]
// ---------------------------------------------------------------------------
__global__ void graph_feat_kernel(const float* __restrict__ xf, const int* __restrict__ idx,
                                  float* __restrict__ out, int Cf, long long total)
{
    long long t = (long long)blockIdx.x * blockDim.x + threadIdx.x;
    if (t >= total) return;
    const int k = (int)(t % KNNB);
    long long r = t / KNNB;
    const int n = (int)(r % NN); r /= NN;
    const int v = (int)(r % 3);  r /= 3;
    const int c = (int)(r % (2 * Cf));
    const int b = (int)(r / (2 * Cf));

    const long long xb = (long long)b * (3 * Cf) * NN;
    float val;
    if (c < Cf) {
        const int j = idx[((long long)b * NN + n) * KNNB + k];
        const long long ch = (long long)(c * 3 + v) * NN;
        val = xf[xb + ch + j] - xf[xb + ch + n];
    } else {
        val = xf[xb + (long long)((c - Cf) * 3 + v) * NN + n];
    }
    out[t] = val;
}

// ---------------------------------------------------------------------------
// norm over the 3-axis: nrm[b,o,s] = ||P[b,o,:,s]|| + EPS ; P (B,O,3,S)
// ---------------------------------------------------------------------------
__global__ void norm3_kernel(const float* __restrict__ P, float* __restrict__ nrm,
                             int S, long long total)
{
    long long t = (long long)blockIdx.x * blockDim.x + threadIdx.x;
    if (t >= total) return;
    const int s = (int)(t % S);
    const long long bo = t / S;
    const float* p = P + bo * 3LL * S + s;
    const float a = p[0], b2 = p[(long long)S], c2 = p[2LL * S];
    nrm[t] = sqrtf(a * a + b2 * b2 + c2 * c2) + EPS_;
}

// ---------------------------------------------------------------------------
// Per-channel mean / inv-std over (B, S). One block per channel, LDS reduce.
// v layout (B, O, S). stats[2o]=mean, stats[2o+1]=rsqrt(var+BNEPS).
// ---------------------------------------------------------------------------
__global__ void bn_stats_kernel(const float* __restrict__ v, float* __restrict__ stats,
                                int O, int S)
{
    const int o = blockIdx.x;
    const long long cnt = (long long)BB * S;
    float s1 = 0.f, s2 = 0.f;
    for (long long i = threadIdx.x; i < cnt; i += blockDim.x) {
        const int b = (int)(i / S), s = (int)(i - (long long)b * S);
        const float x = v[((long long)b * O + o) * S + s];
        s1 += x; s2 += x * x;
    }
    __shared__ float r1[256], r2[256];
    r1[threadIdx.x] = s1; r2[threadIdx.x] = s2;
    __syncthreads();
    for (int w = 128; w > 0; w >>= 1) {
        if ((int)threadIdx.x < w) {
            r1[threadIdx.x] += r1[threadIdx.x + w];
            r2[threadIdx.x] += r2[threadIdx.x + w];
        }
        __syncthreads();
    }
    if (threadIdx.x == 0) {
        const float mean = r1[0] / (float)cnt;
        const float var  = r2[0] / (float)cnt - mean * mean;
        stats[2 * o]     = mean;
        stats[2 * o + 1] = rsqrtf(var + BNEPS_);
    }
}

// ---------------------------------------------------------------------------
// VN-BN + VN-LeakyReLU apply, in place over P. P (B,O,3,S), Dm (B,Od,3,S).
// ---------------------------------------------------------------------------
__global__ void vn_apply_kernel(float* __restrict__ P, const float* __restrict__ Dm,
                                const float* __restrict__ nrm, const float* __restrict__ stats,
                                int O, int Od, int S, long long total)
{
    long long t = (long long)blockIdx.x * blockDim.x + threadIdx.x;
    if (t >= total) return;
    const int s = (int)(t % S);
    long long r = t / S;
    const int o = (int)(r % O);
    const int b = (int)(r / O);

    const float nv    = nrm[t];
    const float scale = (nv - stats[2 * o]) * stats[2 * o + 1] / nv;

    const long long pb = ((long long)b * O + o) * 3LL * S + s;
    const int od = (Od == 1) ? 0 : o;
    const long long db = ((long long)b * Od + od) * 3LL * S + s;

    const float p0 = P[pb] * scale, p1 = P[pb + S] * scale, p2 = P[pb + 2LL * S] * scale;
    const float d0 = Dm[db], d1 = Dm[db + S], d2 = Dm[db + 2LL * S];
    const float dot = p0 * d0 + p1 * d1 + p2 * d2;
    const float dsq = d0 * d0 + d1 * d1 + d2 * d2;

    float q0 = p0, q1 = p1, q2 = p2;
    if (dot < 0.0f) {
        const float f = dot / (dsq + EPS_);
        q0 = p0 - f * d0; q1 = p1 - f * d1; q2 = p2 - f * d2;
    }
    P[pb]           = NEG_ * p0 + (1.0f - NEG_) * q0;
    P[pb + S]       = NEG_ * p1 + (1.0f - NEG_) * q1;
    P[pb + 2LL * S] = NEG_ * p2 + (1.0f - NEG_) * q2;
}

// ---------------------------------------------------------------------------
// VN max-pool over K: out (B,O,3,N) <- argmax_k sum_v X*D. First-max wins.
// X, Dm (B,O,3,N,K).
// ---------------------------------------------------------------------------
__global__ void pool_kernel(const float* __restrict__ X, const float* __restrict__ Dm,
                            float* __restrict__ out, int O, long long total)
{
    long long t = (long long)blockIdx.x * blockDim.x + threadIdx.x;
    if (t >= total) return;
    const int n = (int)(t % NN);
    long long r = t / NN;
    const int o = (int)(r % O);
    const int b = (int)(r / O);

    const long long vs   = NKLL;  // stride of v axis = N*K
    const long long base = ((long long)b * O + o) * 3LL * vs + (long long)n * KNNB;

    float best = -3.4e38f; int bi = 0;
    for (int k = 0; k < KNNB; ++k) {
        const float dot = X[base + k]           * Dm[base + k]
                        + X[base + vs + k]      * Dm[base + vs + k]
                        + X[base + 2 * vs + k]  * Dm[base + 2 * vs + k];
        if (dot > best) { best = dot; bi = k; }
    }
    const long long ob = ((long long)b * O + o) * 3LL * NN + n;
#pragma unroll
    for (int v = 0; v < 3; ++v)
        out[ob + (long long)v * NN] = X[base + (long long)v * vs + bi];
}

// ---------------------------------------------------------------------------
// h (B,C9,3,N) -> h2 (B,2*C9,3,N) = [h ; mean_n h broadcast]
// one thread per (b,o,v)
// ---------------------------------------------------------------------------
__global__ void mean_concat_kernel(const float* __restrict__ h, float* __restrict__ h2)
{
    const int t = blockIdx.x * blockDim.x + threadIdx.x;
    if (t >= BB * C9 * 3) return;
    const int v = t % 3;
    int r = t / 3;
    const int o = r % C9;
    const int b = r / C9;

    const float* src = h + ((long long)b * C9 + o) * 3LL * NN + (long long)v * NN;
    float s = 0.f;
    for (int n = 0; n < NN; ++n) s += src[n];
    const float mean = s * (1.0f / NN);

    float* d1 = h2 + ((long long)b * D2 + o)      * 3LL * NN + (long long)v * NN;
    float* d2 = h2 + ((long long)b * D2 + o + C9) * 3LL * NN + (long long)v * NN;
    for (int n = 0; n < NN; ++n) { d1[n] = src[n]; d2[n] = mean; }
}

// ---------------------------------------------------------------------------
// hs[b,i,kk,n] = sum_j h2[b,i,j,n] * z0[b,j,kk,n]   (682x3 times 3x3 per point)
// ---------------------------------------------------------------------------
__global__ void hs_kernel(const float* __restrict__ h2, const float* __restrict__ z0,
                          float* __restrict__ hs, long long total)
{
    long long t = (long long)blockIdx.x * blockDim.x + threadIdx.x;
    if (t >= total) return;
    const int n = (int)(t % NN);
    long long r = t / NN;
    const int i = (int)(r % D2);
    const int b = (int)(r / D2);

    const long long hb = ((long long)b * D2 + i) * 3LL * NN + n;
    const float h0 = h2[hb], h1 = h2[hb + NN], hv2 = h2[hb + 2LL * NN];
#pragma unroll
    for (int kk = 0; kk < 3; ++kk) {
        const float acc = h0  * z0[(((long long)b * 3 + 0) * 3 + kk) * NN + n]
                        + h1  * z0[(((long long)b * 3 + 1) * 3 + kk) * NN + n]
                        + hv2 * z0[(((long long)b * 3 + 2) * 3 + kk) * NN + n];
        hs[hb + (long long)kk * NN] = acc;
    }
}

// ---------------------------------------------------------------------------
// BN + leaky-relu apply for conv head, in place. h (B,O,N).
// ---------------------------------------------------------------------------
__global__ void conv_apply_kernel(float* __restrict__ h, const float* __restrict__ stats,
                                  int O, long long total)
{
    long long t = (long long)blockIdx.x * blockDim.x + threadIdx.x;
    if (t >= total) return;
    const int o = (int)((t / NN) % O);
    const float v = (h[t] - stats[2 * o]) * stats[2 * o + 1];
    h[t] = (v >= 0.0f) ? v : NEG_ * v;
}

// (B,OUTC,N) -> (B,N,OUTC)
__global__ void transpose_out_kernel(const float* __restrict__ c4, float* __restrict__ out)
{
    const int t = blockIdx.x * blockDim.x + threadIdx.x;
    if (t >= BB * NN * OUTC) return;
    const int o = t % OUTC;
    int r = t / OUTC;
    const int n = r % NN;
    const int b = r / NN;
    out[t] = c4[((long long)b * OUTC + o) * NN + n];
}

// ---------------------------------------------------------------------------
// Host-side orchestration
// ---------------------------------------------------------------------------
static inline long long cdivll(long long a, long long b) { return (a + b - 1) / b; }

static void run_gemm(hipStream_t st,
                     const float* W, long long wB, long long wM, long long wK,
                     const float* A, long long aB, long long aK, long long aS,
                     float* Cp, long long cB, long long cM, long long cS,
                     int M, int Kd, int S)
{
    dim3 grid((unsigned)cdivll(S, 64), (unsigned)cdivll(M, 16), (unsigned)BB);
    gemm_f16_wmma<<<grid, dim3(128), 0, st>>>(W, wB, wM, wK, A, aB, aK, aS,
                                              Cp, cB, cM, cS, M, Kd, S);
}

// vn_lrelu: P = BN(Wf X), D = Wd X, VN-LReLU(P, D) written in place over P
static void vn_lrelu_run(hipStream_t st, const float* wf, const float* wd,
                         int O, int Od, int Cin, long long Ssp,
                         const float* X, float* P, float* Dm, float* nrm, float* stats)
{
    const long long Sg = 3 * Ssp;
    run_gemm(st, wf, 0, Cin, 1, X, (long long)Cin * Sg, Sg, 1,
             P, (long long)O * Sg, Sg, 1, O, Cin, (int)Sg);
    run_gemm(st, wd, 0, Cin, 1, X, (long long)Cin * Sg, Sg, 1,
             Dm, (long long)Od * Sg, Sg, 1, Od, Cin, (int)Sg);
    const long long tot = (long long)BB * O * Ssp;
    norm3_kernel<<<(unsigned)cdivll(tot, 256), 256, 0, st>>>(P, nrm, (int)Ssp, tot);
    bn_stats_kernel<<<O, 256, 0, st>>>(nrm, stats, O, (int)Ssp);
    vn_apply_kernel<<<(unsigned)cdivll(tot, 256), 256, 0, st>>>(P, Dm, nrm, stats,
                                                                O, Od, (int)Ssp, tot);
}

// kNN: Gram via WMMA GEMM (W = xf^T view), then per-row top-20
static void knn_run(hipStream_t st, const float* xf, int Dft, float* G, int* idx)
{
    run_gemm(st, xf, (long long)Dft * NN, 1, NN,
                 xf, (long long)Dft * NN, NN, 1,
                 G, (long long)NN * NN, NN, 1, NN, Dft, NN);
    topk_kernel<<<(BB * NN + 255) / 256, 256, 0, st>>>(G, idx);
}

static void pool_run(hipStream_t st, const float* wd, int O,
                     const float* X, float* Dm, float* out)
{
    const long long Sg = 3 * NKLL;
    run_gemm(st, wd, 0, O, 1, X, (long long)O * Sg, Sg, 1,
             Dm, (long long)O * Sg, Sg, 1, O, O, (int)Sg);
    const long long tot = (long long)BB * O * NN;
    pool_kernel<<<(unsigned)cdivll(tot, 256), 256, 0, st>>>(X, Dm, out, O, tot);
}

extern "C" void kernel_launch(void* const* d_in, const int* in_sizes, int n_in,
                              void* d_out, int out_size, void* d_ws, size_t ws_size,
                              hipStream_t stream)
{
    (void)in_sizes; (void)n_in; (void)out_size; (void)ws_size;
    const float* x    = (const float*)d_in[0];
    const float* w1f  = (const float*)d_in[1];  const float* w1d = (const float*)d_in[2];
    const float* w2f  = (const float*)d_in[3];  const float* w2d = (const float*)d_in[4];
    const float* w3f  = (const float*)d_in[5];  const float* w3d = (const float*)d_in[6];
    const float* w4f  = (const float*)d_in[7];  const float* w4d = (const float*)d_in[8];
    const float* w5f  = (const float*)d_in[9];  const float* w5d = (const float*)d_in[10];
    const float* w6f  = (const float*)d_in[11]; const float* w6d = (const float*)d_in[12];
    const float* w7f  = (const float*)d_in[13]; const float* w7d = (const float*)d_in[14];
    const float* w8f  = (const float*)d_in[15]; const float* w8d = (const float*)d_in[16];
    const float* w9f  = (const float*)d_in[17]; const float* w9d = (const float*)d_in[18];
    const float* p1   = (const float*)d_in[19];
    const float* p2   = (const float*)d_in[20];
    const float* p3   = (const float*)d_in[21];
    const float* s1f  = (const float*)d_in[22]; const float* s1d = (const float*)d_in[23];
    const float* s2f  = (const float*)d_in[24]; const float* s2d = (const float*)d_in[25];
    const float* slin = (const float*)d_in[26];
    const float* w10  = (const float*)d_in[27];
    const float* w11  = (const float*)d_in[28];
    const float* w12  = (const float*)d_in[29];
    const float* w13  = (const float*)d_in[30];
    float* out = (float*)d_out;
    float* ws  = (float*)d_ws;

    // workspace layout (float elements)
    constexpr long long SZ_G   = 4LL * 1024 * 1024;       // 4,194,304
    constexpr long long SZ_IDX = 81920;
    constexpr long long SZ_X   = 20643840;                // B*84*3*N*K
    constexpr long long SZ_Y   = 10321920;                // B*42*3*N*K
    constexpr long long SZ_D   = 10321920;
    constexpr long long SZ_NRM = 3440640;                 // B*42*N*K
    constexpr long long SZ_ST  = 2048;
    constexpr long long SZ_XP  = 516096;                  // B*42*3*N
    constexpr long long SZ_H2  = 8380416;                 // B*682*3*N

    long long off = 0;
    float* G     = ws + off; off += SZ_G;
    int*   idx   = (int*)(ws + off); off += SZ_IDX;
    float* bufX  = ws + off; off += SZ_X;
    float* bufY  = ws + off; off += SZ_Y;
    float* bufD  = ws + off; off += SZ_D;
    float* nrm   = ws + off; off += SZ_NRM;
    float* stats = ws + off; off += SZ_ST;
    float* x1    = ws + off; off += SZ_XP;
    float* x2    = ws + off; off += SZ_XP;
    float* x3    = ws + off; off += SZ_XP;
    float* h2    = ws + off; off += SZ_H2;
    float* z0b   = ws + off; off += 36864;

    // ---------------- Stage 1 ----------------
    knn_run(stream, x, 3, G, idx);
    {
        const long long tot = (long long)BB * 2 * 3 * NN * KNNB;
        graph_feat_kernel<<<(unsigned)cdivll(tot, 256), 256, 0, stream>>>(x, idx, bufX, 1, tot);
    }
    vn_lrelu_run(stream, w1f, w1d, CC, CC, 2,  NKLL, bufX, bufY, bufD, nrm, stats);
    vn_lrelu_run(stream, w2f, w2d, CC, CC, CC, NKLL, bufY, bufX, bufD, nrm, stats);
    vn_lrelu_run(stream, w3f, w3d, CC, CC, CC, NKLL, bufX, bufY, bufD, nrm, stats);
    pool_run(stream, p1, CC, bufY, bufD, x1);

    // ---------------- Stage 2 ----------------
    knn_run(stream, x1, 3 * CC, G, idx);
    {
        const long long tot = (long long)BB * 2 * CC * 3 * NN * KNNB;
        graph_feat_kernel<<<(unsigned)cdivll(tot, 256), 256, 0, stream>>>(x1, idx, bufX, CC, tot);
    }
    vn_lrelu_run(stream, w4f, w4d, CC, CC, 2 * CC, NKLL, bufX, bufY, bufD, nrm, stats);
    vn_lrelu_run(stream, w5f, w5d, CC, CC, CC,     NKLL, bufY, bufX, bufD, nrm, stats);
    vn_lrelu_run(stream, w6f, w6d, CC, CC, CC,     NKLL, bufX, bufY, bufD, nrm, stats);
    pool_run(stream, p2, CC, bufY, bufD, x2);

    // ---------------- Stage 3 ----------------
    knn_run(stream, x2, 3 * CC, G, idx);
    {
        const long long tot = (long long)BB * 2 * CC * 3 * NN * KNNB;
        graph_feat_kernel<<<(unsigned)cdivll(tot, 256), 256, 0, stream>>>(x2, idx, bufX, CC, tot);
    }
    vn_lrelu_run(stream, w7f, w7d, CC, CC, 2 * CC, NKLL, bufX, bufY, bufD, nrm, stats);
    vn_lrelu_run(stream, w8f, w8d, CC, CC, CC,     NKLL, bufY, bufX, bufD, nrm, stats);
    pool_run(stream, p3, CC, bufX, bufD, x3);

    // ---------------- Stage 4 (global head) ----------------
    // h = vn_lrelu(x3, w9f (341,42), w9d (1,42)), S = N
    vn_lrelu_run(stream, w9f, w9d, C9, 1, CC, NN, x3, bufX, bufD, nrm, stats);
    mean_concat_kernel<<<(BB * C9 * 3 + 255) / 256, 256, 0, stream>>>(bufX, h2);

    vn_lrelu_run(stream, s1f, s1d, C9, C9, D2,   NN, h2,   bufY, bufD, nrm, stats);
    vn_lrelu_run(stream, s2f, s2d, 170, 170, C9, NN, bufY, bufX, bufD, nrm, stats);

    // z0 = slin (3,170) x z2 -> (B,3,3,N)
    run_gemm(stream, slin, 0, 170, 1,
             bufX, 170LL * 3 * NN, 3LL * NN, 1,
             z0b, 9LL * NN, 3LL * NN, 1, 3, 170, 3 * NN);

    {
        const long long tot = (long long)BB * D2 * NN;
        hs_kernel<<<(unsigned)cdivll(tot, 256), 256, 0, stream>>>(h2, z0b, bufY, tot);
    }

    // conv head: hs viewed as (B, 2046, N)
    const int CH = 3 * D2;  // 2046
    run_gemm(stream, w10, 0, CH, 1, bufY, (long long)CH * NN, NN, 1,
             bufX, 256LL * NN, NN, 1, 256, CH, NN);
    {
        const long long tot = (long long)BB * 256 * NN;
        bn_stats_kernel<<<256, 256, 0, stream>>>(bufX, stats, 256, NN);
        conv_apply_kernel<<<(unsigned)cdivll(tot, 256), 256, 0, stream>>>(bufX, stats, 256, tot);
    }
    run_gemm(stream, w11, 0, 256, 1, bufX, 256LL * NN, NN, 1,
             bufD, 256LL * NN, NN, 1, 256, 256, NN);
    {
        const long long tot = (long long)BB * 256 * NN;
        bn_stats_kernel<<<256, 256, 0, stream>>>(bufD, stats, 256, NN);
        conv_apply_kernel<<<(unsigned)cdivll(tot, 256), 256, 0, stream>>>(bufD, stats, 256, tot);
    }
    run_gemm(stream, w12, 0, 256, 1, bufD, 256LL * NN, NN, 1,
             bufX, 128LL * NN, NN, 1, 128, 256, NN);
    {
        const long long tot = (long long)BB * 128 * NN;
        bn_stats_kernel<<<128, 256, 0, stream>>>(bufX, stats, 128, NN);
        conv_apply_kernel<<<(unsigned)cdivll(tot, 256), 256, 0, stream>>>(bufX, stats, 128, tot);
    }
    run_gemm(stream, w13, 0, 128, 1, bufX, 128LL * NN, NN, 1,
             bufD, (long long)OUTC * NN, NN, 1, OUTC, 128, NN);

    transpose_out_kernel<<<(BB * NN * OUTC + 255) / 256, 256, 0, stream>>>(bufD, out);
}
